// ParamReadout_26414048870994
// MI455X (gfx1250) — matrix operand order repaired and checked by reading, not compiled, not run on hardware
//
#include <hip/hip_runtime.h>
#include <math.h>

typedef __attribute__((ext_vector_type(16))) __bf16 v16bf;
typedef __attribute__((ext_vector_type(8)))  float  v8f;

#define WAVES_PER_BLOCK 8
#define ROWS_PER_WAVE   16
#define ROWS_PER_BLOCK  (WAVES_PER_BLOCK * ROWS_PER_WAVE)

__device__ __forceinline__ __bf16 f2bf(float x) { return (__bf16)x; }

// CDNA5 has a hardware V_TANH_F32 transcendental; use it if clang exposes it.
__device__ __forceinline__ float fast_tanh(float x) {
#if __has_builtin(__builtin_amdgcn_tanhf)
    return __builtin_amdgcn_tanhf(x);
#else
    return tanhf(x);
#endif
}

union AFrag { v16bf v; uint4 q[2]; };

// MODE 0: atom   (row = h[r])
// MODE 1: bond   (row = h[idxA[2r]] + h[idxA[2r+1]])
// MODE 2: angle  (row = h[idxA[r]] + h[idxB[2r]] + h[idxB[2r+1]])
// MODE 3: torsion(row = h[idxA[4r+2]] + h[idxA[4r+3]])
template <int MODE>
__global__ __launch_bounds__(256) void mlp_kernel(
    const float* __restrict__ h,
    const int*   __restrict__ idxA,
    const int*   __restrict__ idxB,
    const float* __restrict__ W1,   // [128][32] row-major (D x R)
    const float* __restrict__ b1,   // [32]
    const float* __restrict__ W2,   // [32][2]   row-major (R x 2)
    const float* __restrict__ b2,   // [2]
    float* __restrict__ out0,
    float* __restrict__ out1,
    int n_rows)
{
    __shared__ __align__(16) __bf16 sW1t[32][128];                 // W1 transposed (N x K), bf16
    __shared__ float sW2[32][2];
    __shared__ float sb1[32];
    __shared__ float sb2[2];
    __shared__ __align__(16) __bf16 sA[WAVES_PER_BLOCK][16][128];  // input tiles, bf16
    __shared__ float sAct[WAVES_PER_BLOCK][16][32];                // tanh activations

    const int tid = threadIdx.x;

    // ---- stage weights into LDS (transpose W1, convert to bf16) ----
    for (int i = tid; i < 128 * 32; i += 256) {
        const int k = i >> 5;     // 0..127
        const int n = i & 31;     // 0..31
        sW1t[n][k] = f2bf(W1[i]);
    }
    if (tid < 64) sW2[tid >> 1][tid & 1] = W2[tid];
    if (tid < 32) sb1[tid] = b1[tid];
    if (tid < 2)  sb2[tid] = b2[tid];
    __syncthreads();

    const int wave = tid >> 5;
    const int lane = tid & 31;
    const int row_base = blockIdx.x * ROWS_PER_BLOCK + wave * ROWS_PER_WAVE;

    // ---- gather + sum + convert: lane owns float4 chunk c4 = lane, loops 16 rows ----
    const float4* __restrict__ h4 = (const float4*)h;
    const int c4 = lane;  // 0..31  (32 float4 per 128-float row)
    for (int r = 0; r < ROWS_PER_WAVE; ++r) {
        int row = row_base + r;
        if (row >= n_rows) row = n_rows - 1;   // clamp (tail rows discarded at store)
        float4 s;
        if (MODE == 0) {
            s = h4[(long)row * 32 + c4];
        } else if (MODE == 1) {
            const int i0 = idxA[2 * row], i1 = idxA[2 * row + 1];
            const float4 a = h4[(long)i0 * 32 + c4];
            const float4 b = h4[(long)i1 * 32 + c4];
            s = make_float4(a.x + b.x, a.y + b.y, a.z + b.z, a.w + b.w);
        } else if (MODE == 2) {
            const int ic = idxA[row];
            const int i0 = idxB[2 * row], i1 = idxB[2 * row + 1];
            const float4 a = h4[(long)ic * 32 + c4];
            const float4 b = h4[(long)i0 * 32 + c4];
            const float4 c = h4[(long)i1 * 32 + c4];
            s = make_float4(a.x + b.x + c.x, a.y + b.y + c.y,
                            a.z + b.z + c.z, a.w + b.w + c.w);
        } else {
            const int i0 = idxA[4 * row + 2], i1 = idxA[4 * row + 3];
            const float4 a = h4[(long)i0 * 32 + c4];
            const float4 b = h4[(long)i1 * 32 + c4];
            s = make_float4(a.x + b.x, a.y + b.y, a.z + b.z, a.w + b.w);
        }
        __bf16* dst = &sA[wave][r][c4 * 4];
        dst[0] = f2bf(s.x); dst[1] = f2bf(s.y);
        dst[2] = f2bf(s.z); dst[3] = f2bf(s.w);
    }
    __syncthreads();

    // ---- GEMM1: (16x128) @ (128x32) via v_wmma_f32_16x16x32_bf16 ----
    // A frag (16-bit A 16x32): lane = M + 16*half; K = kk*32 + half*16 + [0..15]
    // B frag (16-bit B 32x16): lane = N + 16*half; K = kk*32 + half*16 + [0..15]
    const int M    = lane & 15;
    const int half = lane >> 4;

    AFrag a[4], bf0[4], bf1[4];
    const uint4* __restrict__ pa  = (const uint4*)&sA[wave][M][0];   // 16 x uint4 / row
    const uint4* __restrict__ pb0 = (const uint4*)&sW1t[M][0];       // N = M       (tile 0)
    const uint4* __restrict__ pb1 = (const uint4*)&sW1t[16 + M][0];  // N = 16 + M  (tile 1)
#pragma unroll
    for (int kk = 0; kk < 4; ++kk) {
        const int kA = kk * 32 + half * 16;       // bf16 index; /8 -> uint4 index
        a[kk].q[0] = pa[kA >> 3];
        a[kk].q[1] = pa[(kA >> 3) + 1];
        const int kB = kk * 32 + half * 16;
        bf0[kk].q[0] = pb0[kB >> 3];
        bf0[kk].q[1] = pb0[(kB >> 3) + 1];
        bf1[kk].q[0] = pb1[kB >> 3];
        bf1[kk].q[1] = pb1[(kB >> 3) + 1];
    }

    v8f acc0 = {}; v8f acc1 = {};
#pragma unroll
    for (int kk = 0; kk < 4; ++kk) {
        acc0 = __builtin_amdgcn_wmma_f32_16x16x32_bf16(
            false, a[kk].v, false, bf0[kk].v, (short)0, acc0, false, false);
        acc1 = __builtin_amdgcn_wmma_f32_16x16x32_bf16(
            false, a[kk].v, false, bf1[kk].v, (short)0, acc1, false, false);
    }

    // ---- bias + tanh; C layout: VGPR v -> row M' = v + 8*half, col N = lane&15 ----
    const float bias0 = sb1[M];
    const float bias1 = sb1[16 + M];
#pragma unroll
    for (int v = 0; v < 8; ++v) {
        const int rowm = v + 8 * half;
        sAct[wave][rowm][M]      = fast_tanh(acc0[v] + bias0);
        sAct[wave][rowm][16 + M] = fast_tanh(acc1[v] + bias1);
    }
    __syncthreads();

    // ---- GEMM2: (16x32) @ (32x2) + b2, abs, store ----
    // lanes 0-15 -> col 0 rows 0-15 (coalesced out0); lanes 16-31 -> col 1 (coalesced out1)
    const int r2 = lane & 15;
    const int c2 = lane >> 4;
    float sum = sb2[c2];
#pragma unroll
    for (int r = 0; r < 32; ++r) sum += sAct[wave][r2][r] * sW2[r][c2];
    const float val = fabsf(sum);
    const int row = row_base + r2;
    if (row < n_rows) {
        (c2 == 0 ? out0 : out1)[row] = val;
    }
}

// sigma = mean(k[a], k[b]); eps = sqrt(eq[a]*eq[b])
__global__ __launch_bounds__(256) void pair_kernel(
    const float* __restrict__ k_atom,
    const float* __restrict__ eq_atom,
    const int*   __restrict__ idx,
    float* __restrict__ sigma,
    float* __restrict__ eps,
    int n)
{
    const int i = blockIdx.x * blockDim.x + threadIdx.x;
    if (i < n) {
        const int a = idx[2 * i];
        const int b = idx[2 * i + 1];
        sigma[i] = 0.5f * (k_atom[a] + k_atom[b]);
        eps[i]   = sqrtf(eq_atom[a] * eq_atom[b]);
    }
}

extern "C" void kernel_launch(void* const* d_in, const int* in_sizes, int n_in,
                              void* d_out, int out_size, void* d_ws, size_t ws_size,
                              hipStream_t stream) {
    const float* h    = (const float*)d_in[0];
    const int*   bond = (const int*)d_in[1];
    const int*   angc = (const int*)d_in[2];
    const int*   angs = (const int*)d_in[3];
    const int*   tors = (const int*)d_in[4];
    const int*   p14  = (const int*)d_in[5];
    const int*   pnb  = (const int*)d_in[6];
    const float* W1a = (const float*)d_in[7];
    const float* b1a = (const float*)d_in[8];
    const float* W2a = (const float*)d_in[9];
    const float* b2a = (const float*)d_in[10];
    const float* W1b = (const float*)d_in[11];
    const float* b1b = (const float*)d_in[12];
    const float* W2b = (const float*)d_in[13];
    const float* b2b = (const float*)d_in[14];

    const int n_atom  = in_sizes[0] / 128;
    const int n_bond  = in_sizes[1] / 2;
    const int n_angle = in_sizes[2];
    const int n_tor   = in_sizes[4] / 4;
    const int n_14    = in_sizes[5] / 2;
    const int n_nb    = in_sizes[6] / 2;

    float* out     = (float*)d_out;
    float* k_atom  = out;
    float* eq_atom = k_atom + n_atom;
    float* bond0   = eq_atom + n_atom;
    float* bond1   = bond0 + n_bond;
    float* ang0    = bond1 + n_bond;
    float* ang1    = ang0 + n_angle;
    float* tor0    = ang1 + n_angle;
    float* tor1    = tor0 + n_tor;
    float* s14     = tor1 + n_tor;
    float* e14     = s14 + n_14;
    float* snb     = e14 + n_14;
    float* enb     = snb + n_nb;

    const dim3 blk(256);
    const auto gblocks = [](int n) { return dim3((unsigned)((n + ROWS_PER_BLOCK - 1) / ROWS_PER_BLOCK)); };

    mlp_kernel<0><<<gblocks(n_atom),  blk, 0, stream>>>(h, nullptr, nullptr,
        W1a, b1a, W2a, b2a, k_atom, eq_atom, n_atom);
    mlp_kernel<1><<<gblocks(n_bond),  blk, 0, stream>>>(h, bond, nullptr,
        W1b, b1b, W2b, b2b, bond0, bond1, n_bond);
    mlp_kernel<2><<<gblocks(n_angle), blk, 0, stream>>>(h, angc, angs,
        W1b, b1b, W2b, b2b, ang0, ang1, n_angle);
    mlp_kernel<3><<<gblocks(n_tor),   blk, 0, stream>>>(h, tors, nullptr,
        W1b, b1b, W2b, b2b, tor0, tor1, n_tor);

    // pair mixing reads k_atom/eq_atom written above (same-stream ordering)
    pair_kernel<<<dim3((unsigned)((n_14 + 255) / 256)), blk, 0, stream>>>(
        k_atom, eq_atom, p14, s14, e14, n_14);
    pair_kernel<<<dim3((unsigned)((n_nb + 255) / 256)), blk, 0, stream>>>(
        k_atom, eq_atom, pnb, snb, enb, n_nb);
}
